// DeformableTransformer_75892072120613
// MI455X (gfx1250) — matrix operand order repaired and checked by reading, not compiled, not run on hardware
//
#include <hip/hip_runtime.h>
#include <hip/hip_bf16.h>

#define S_TOTAL_C 17821
#define NB 2
#define TOK (NB * S_TOTAL_C)   /* 35642 */
#define TPAD 35648             /* 64 * 557 */
#define DMODEL 256
#define NHEADS 8
#define DHEAD 32
#define LVLS 4
#define PTS 4
#define DFFN 1024
#define NLAYERS 6
#define FFN_CHUNK 8192

typedef __attribute__((ext_vector_type(16))) _Float16 v16h;
typedef __attribute__((ext_vector_type(8)))  _Float16 v8h;
typedef __attribute__((ext_vector_type(8)))  float    v8f;

// ---------------------------------------------------------------------------
// Tiled GEMM: out[M,Nout] = (A (+A2)) @ W + bias, optional ReLU.
// 256 threads = 8 waves; block tile 64(M) x 128(N); K stepped by 32 via LDS.
// Each wave: 16x64 output = 4 WMMAs per K-step, A fragment reused 4x.
// Fragment layouts: contiguous k order (k = 16*(lane/16) + s), same for A and B.
// ---------------------------------------------------------------------------
__global__ __launch_bounds__(256)
void gemm_wmma_kernel(const float* __restrict__ A, const float* __restrict__ A2,
                      const float* __restrict__ W, const float* __restrict__ bias,
                      float* __restrict__ out, int Mvalid, int K, int Nout,
                      int relu, int hasAdd)
{
    __shared__ __align__(16) _Float16 Ash[64 * 32];    // [row][k]
    __shared__ __align__(16) _Float16 Bsh[128 * 32];   // [col][k] (W transposed)

    const int t    = threadIdx.x;
    const int lane = t & 31;
    const int wave = t >> 5;
    const int wm   = wave & 3;    // 4 row sub-tiles of 16
    const int wq   = wave >> 2;   // 2 col groups of 64
    const int l16  = lane & 15;
    const bool hi  = lane >= 16;
    const int rowBase = blockIdx.x * 64;
    const int colBase = blockIdx.y * 128;

    // staging indices (hoisted out of the K loop)
    const int ar   = t >> 2;            // 0..63  A row
    const int aseg = (t & 3) * 8;       // 0/8/16/24
    const int agr  = rowBase + ar;
    const int bc_s = t >> 1;            // 0..127 B col
    const int bks  = (t & 1) * 16;      // 0/16
    const int bgc  = colBase + bc_s;

    v8f acc0 = {}, acc1 = {}, acc2 = {}, acc3 = {};

    for (int k0 = 0; k0 < K; k0 += 32) {
        __syncthreads();
        // ---- stage A tile (64x32), fused add, fp32 -> f16, one b128 store
        {
            float f0 = 0.f, f1 = 0.f, f2 = 0.f, f3 = 0.f;
            float f4 = 0.f, f5 = 0.f, f6 = 0.f, f7 = 0.f;
            if (agr < Mvalid) {
                const float4* ap = (const float4*)(A + (size_t)agr * K + k0 + aseg);
                float4 u = ap[0], v = ap[1];
                f0 = u.x; f1 = u.y; f2 = u.z; f3 = u.w;
                f4 = v.x; f5 = v.y; f6 = v.z; f7 = v.w;
                if (hasAdd) {
                    const float4* ap2 = (const float4*)(A2 + (size_t)agr * K + k0 + aseg);
                    float4 u2 = ap2[0], v2 = ap2[1];
                    f0 += u2.x; f1 += u2.y; f2 += u2.z; f3 += u2.w;
                    f4 += v2.x; f5 += v2.y; f6 += v2.z; f7 += v2.w;
                }
            }
            v8h hv;
            hv[0] = (_Float16)f0; hv[1] = (_Float16)f1;
            hv[2] = (_Float16)f2; hv[3] = (_Float16)f3;
            hv[4] = (_Float16)f4; hv[5] = (_Float16)f5;
            hv[6] = (_Float16)f6; hv[7] = (_Float16)f7;
            *(v8h*)&Ash[ar * 32 + aseg] = hv;
        }
        // ---- stage W tile transposed (128 cols x 32 k), two b128 stores
        {
            const float* wp = W + (size_t)(k0 + bks) * Nout + bgc;
            v8h b0, b1;
            #pragma unroll
            for (int j = 0; j < 8; ++j)
                b0[j] = (_Float16)wp[(size_t)j * Nout];
            #pragma unroll
            for (int j = 0; j < 8; ++j)
                b1[j] = (_Float16)wp[(size_t)(j + 8) * Nout];
            *(v8h*)&Bsh[bc_s * 32 + bks]     = b0;
            *(v8h*)&Bsh[bc_s * 32 + bks + 8] = b1;
        }
        __syncthreads();

        // ---- A fragment: lane<16 -> K 0..15; lane>=16 -> K 16..31
        const int abase = (wm * 16 + l16) * 32 + (hi ? 16 : 0);
        v8h alo = *(const v8h*)&Ash[abase];
        v8h ahi = *(const v8h*)&Ash[abase + 8];
        v16h af = __builtin_shufflevector(alo, ahi,
                                          0, 1, 2, 3, 4, 5, 6, 7,
                                          8, 9, 10, 11, 12, 13, 14, 15);
        // ---- B fragments: N = lane&15; lanes 0-15 K 0..15, lanes 16-31 K 16..31
        const int kb   = hi ? 16 : 0;
        const int bcol = wq * 64 + l16;
        v16h bf0 = *(const v16h*)&Bsh[(bcol     ) * 32 + kb];
        v16h bf1 = *(const v16h*)&Bsh[(bcol + 16) * 32 + kb];
        v16h bf2 = *(const v16h*)&Bsh[(bcol + 32) * 32 + kb];
        v16h bf3 = *(const v16h*)&Bsh[(bcol + 48) * 32 + kb];

        acc0 = __builtin_amdgcn_wmma_f32_16x16x32_f16(false, af, false, bf0, (short)0, acc0, false, false);
        acc1 = __builtin_amdgcn_wmma_f32_16x16x32_f16(false, af, false, bf1, (short)0, acc1, false, false);
        acc2 = __builtin_amdgcn_wmma_f32_16x16x32_f16(false, af, false, bf2, (short)0, acc2, false, false);
        acc3 = __builtin_amdgcn_wmma_f32_16x16x32_f16(false, af, false, bf3, (short)0, acc3, false, false);
    }

    // ---- epilogue: C/D layout -> row = v + (hi?8:0), col = lane&15
    const int orow0 = rowBase + wm * 16 + (hi ? 8 : 0);
    const int ocol  = colBase + wq * 64 + l16;
    const float b0 = bias[ocol];
    const float b1 = bias[ocol + 16];
    const float b2 = bias[ocol + 32];
    const float b3 = bias[ocol + 48];
    #pragma unroll
    for (int v = 0; v < 8; ++v) {
        int r = orow0 + v;
        if (r < Mvalid) {
            float f0 = acc0[v] + b0;
            float f1 = acc1[v] + b1;
            float f2 = acc2[v] + b2;
            float f3 = acc3[v] + b3;
            if (relu) {
                f0 = f0 > 0.f ? f0 : 0.f; f1 = f1 > 0.f ? f1 : 0.f;
                f2 = f2 > 0.f ? f2 : 0.f; f3 = f3 > 0.f ? f3 : 0.f;
            }
            float* op = out + (size_t)r * Nout + ocol;
            op[0]  = f0;
            op[16] = f1;
            op[32] = f2;
            op[48] = f3;
        }
    }
}

// ---------------------------------------------------------------------------
// MS-deformable-attention sampling. One block per token, one wave per head,
// one lane per DH channel. Fuses softmax over L*P=16 attention weights,
// reference-point computation, and zero-padded bilinear gathers.
// ---------------------------------------------------------------------------
__device__ __forceinline__
float fetch_val(const float* __restrict__ value, int base, int xi, int yi,
                int Wl, int Hl, int ch)
{
    if (xi >= 0 && xi < Wl && yi >= 0 && yi < Hl)
        return value[(size_t)(base + yi * Wl + xi) * DMODEL + ch];
    return 0.0f;
}

__global__ __launch_bounds__(256)
void ms_sample_kernel(const float* __restrict__ value, const float* __restrict__ so,
                      const float* __restrict__ aw_raw, const float* __restrict__ vr,
                      float* __restrict__ attn, int T)
{
    const int HH[LVLS] = {100, 50, 25, 13};
    const int WW[LVLS] = {134, 67, 34, 17};
    const int ST[LVLS] = {0, 13400, 16750, 17600};

    int token = blockIdx.x;
    if (token >= T) return;
    int h    = threadIdx.x >> 5;   // head
    int lane = threadIdx.x & 31;   // channel within head
    int n = token / S_TOTAL_C;
    int s = token % S_TOTAL_C;

    int ls = (s < 13400) ? 0 : (s < 16750) ? 1 : (s < 17600) ? 2 : 3;
    int rs = s - ST[ls];
    int cy = rs / WW[ls];
    int cx = rs % WW[ls];
    float brx = ((float)cx + 0.5f) / (vr[(n * LVLS + ls) * 2 + 0] * (float)WW[ls]);
    float bry = ((float)cy + 0.5f) / (vr[(n * LVLS + ls) * 2 + 1] * (float)HH[ls]);

    // softmax over 16 weights for this (token, head)
    const float* awp = aw_raw + (size_t)token * (NHEADS * LVLS * PTS) + h * (LVLS * PTS);
    float e[LVLS * PTS];
    float m = -1e30f;
    #pragma unroll
    for (int i = 0; i < LVLS * PTS; ++i) { float a = awp[i]; m = a > m ? a : m; }
    float ssum = 0.0f;
    #pragma unroll
    for (int i = 0; i < LVLS * PTS; ++i) { e[i] = __expf(awp[i] - m); ssum += e[i]; }
    float inv = 1.0f / ssum;

    const float* sop = so + (size_t)token * DMODEL + h * (LVLS * PTS * 2);
    int ch = h * DHEAD + lane;
    float acc = 0.0f;
    #pragma unroll
    for (int l = 0; l < LVLS; ++l) {
        int Wl = WW[l], Hl = HH[l];
        float refx = brx * vr[(n * LVLS + l) * 2 + 0];
        float refy = bry * vr[(n * LVLS + l) * 2 + 1];
        int base = n * S_TOTAL_C + ST[l];
        #pragma unroll
        for (int p = 0; p < PTS; ++p) {
            float ox = sop[(l * PTS + p) * 2 + 0];
            float oy = sop[(l * PTS + p) * 2 + 1];
            // x = (ref + off/W) * W - 0.5
            float x = refx * (float)Wl + ox - 0.5f;
            float y = refy * (float)Hl + oy - 0.5f;
            float fx0 = floorf(x), fy0 = floorf(y);
            int x0 = (int)fx0, y0 = (int)fy0;
            float tx = x - fx0, ty = y - fy0;
            float v00 = fetch_val(value, base, x0,     y0,     Wl, Hl, ch);
            float v10 = fetch_val(value, base, x0 + 1, y0,     Wl, Hl, ch);
            float v01 = fetch_val(value, base, x0,     y0 + 1, Wl, Hl, ch);
            float v11 = fetch_val(value, base, x0 + 1, y0 + 1, Wl, Hl, ch);
            float bil = v00 * (1.0f - tx) * (1.0f - ty) + v10 * tx * (1.0f - ty)
                      + v01 * (1.0f - tx) * ty           + v11 * tx * ty;
            acc += (e[l * PTS + p] * inv) * bil;
        }
    }
    attn[(size_t)token * DMODEL + ch] = acc;
}

// ---------------------------------------------------------------------------
// out = LayerNorm(x + y) * s + b ; one 256-thread block per token.
// ---------------------------------------------------------------------------
__global__ __launch_bounds__(256)
void resid_ln_kernel(const float* __restrict__ x, const float* __restrict__ y,
                     const float* __restrict__ s, const float* __restrict__ b,
                     float* __restrict__ out, int T)
{
    __shared__ float red[256];
    int t = blockIdx.x;
    if (t >= T) return;
    int c = threadIdx.x;
    float a = x[(size_t)t * DMODEL + c] + y[(size_t)t * DMODEL + c];
    red[c] = a;
    __syncthreads();
    for (int off = 128; off > 0; off >>= 1) {
        if (c < off) red[c] += red[c + off];
        __syncthreads();
    }
    float mean = red[0] * (1.0f / (float)DMODEL);
    __syncthreads();
    float d = a - mean;
    red[c] = d * d;
    __syncthreads();
    for (int off = 128; off > 0; off >>= 1) {
        if (c < off) red[c] += red[c + off];
        __syncthreads();
    }
    float var = red[0] * (1.0f / (float)DMODEL);
    out[(size_t)t * DMODEL + c] = d * rsqrtf(var + 1e-5f) * s[c] + b[c];
}

// ---------------------------------------------------------------------------
extern "C" void kernel_launch(void* const* d_in, const int* in_sizes, int n_in,
                              void* d_out, int out_size, void* d_ws, size_t ws_size,
                              hipStream_t stream)
{
    (void)in_sizes; (void)n_in; (void)out_size; (void)ws_size;
    const float* src  = (const float*)d_in[0];
    const float* pos  = (const float*)d_in[1];
    const float* vr   = (const float*)d_in[2];
    const float* so_w = (const float*)d_in[3];
    const float* so_b = (const float*)d_in[4];
    const float* aw_w = (const float*)d_in[5];
    const float* aw_b = (const float*)d_in[6];
    const float* vp_w = (const float*)d_in[7];
    const float* vp_b = (const float*)d_in[8];
    const float* op_w = (const float*)d_in[9];
    const float* op_b = (const float*)d_in[10];
    const float* n1_s = (const float*)d_in[11];
    const float* n1_b = (const float*)d_in[12];
    const float* l1_w = (const float*)d_in[13];
    const float* l1_b = (const float*)d_in[14];
    const float* l2_w = (const float*)d_in[15];
    const float* l2_b = (const float*)d_in[16];
    const float* n2_s = (const float*)d_in[17];
    const float* n2_b = (const float*)d_in[18];

    // workspace carve (floats)
    float* bufA  = (float*)d_ws;                    // so / src2        [TPAD,256]
    float* bufB  = bufA  + (size_t)TPAD * DMODEL;   // value / xm       [TPAD,256]
    float* bufC  = bufB  + (size_t)TPAD * DMODEL;   // attn / ffn_out   [TPAD,256]
    float* bufX  = bufC  + (size_t)TPAD * DMODEL;   // next-layer src   [TPAD,256]
    float* bufAW = bufX  + (size_t)TPAD * DMODEL;   // aw raw           [TPAD,128]
    float* bufH  = bufAW + (size_t)TPAD * 128;      // ffn hidden chunk [FFN_CHUNK,1024]

    const int MT = TPAD / 64;  // 557 M-tiles
    dim3 blk(256);

    const float* x = src;
    for (int l = 0; l < NLAYERS; ++l) {
        // sampling offsets:  (x + pos) @ so_w + so_b   -> bufA  [T,256]
        gemm_wmma_kernel<<<dim3(MT, 2), blk, 0, stream>>>(
            x, pos, so_w + (size_t)l * DMODEL * 256, so_b + (size_t)l * 256,
            bufA, TOK, DMODEL, 256, 0, 1);
        // attention weights: (x + pos) @ aw_w + aw_b   -> bufAW [T,128]
        gemm_wmma_kernel<<<dim3(MT, 1), blk, 0, stream>>>(
            x, pos, aw_w + (size_t)l * DMODEL * 128, aw_b + (size_t)l * 128,
            bufAW, TOK, DMODEL, 128, 0, 1);
        // value projection:   x @ vp_w + vp_b          -> bufB  [T,256]
        gemm_wmma_kernel<<<dim3(MT, 2), blk, 0, stream>>>(
            x, x, vp_w + (size_t)l * DMODEL * DMODEL, vp_b + (size_t)l * DMODEL,
            bufB, TOK, DMODEL, DMODEL, 0, 0);
        // deformable sampling -> bufC [T,256]
        ms_sample_kernel<<<dim3(TOK), blk, 0, stream>>>(bufB, bufA, bufAW, vr, bufC, TOK);
        // output projection: bufC @ op_w + op_b -> bufA (src2)
        gemm_wmma_kernel<<<dim3(MT, 2), blk, 0, stream>>>(
            bufC, bufC, op_w + (size_t)l * DMODEL * DMODEL, op_b + (size_t)l * DMODEL,
            bufA, TOK, DMODEL, DMODEL, 0, 0);
        // LN1: xm = LN(x + src2) -> bufB
        resid_ln_kernel<<<dim3(TOK), blk, 0, stream>>>(
            x, bufA, n1_s + (size_t)l * DMODEL, n1_b + (size_t)l * DMODEL, bufB, TOK);
        // FFN (chunked over rows to bound the 1024-wide hidden buffer)
        for (int c0 = 0; c0 < TPAD; c0 += FFN_CHUNK) {
            int rows = TPAD - c0;
            if (rows > FFN_CHUNK) rows = FFN_CHUNK;
            int mv = TOK - c0;
            if (mv > rows) mv = rows;
            if (mv < 0) mv = 0;
            gemm_wmma_kernel<<<dim3(rows / 64, DFFN / 128), blk, 0, stream>>>(
                bufB + (size_t)c0 * DMODEL, bufB,
                l1_w + (size_t)l * DMODEL * DFFN, l1_b + (size_t)l * DFFN,
                bufH, mv, DMODEL, DFFN, 1, 0);
            gemm_wmma_kernel<<<dim3(rows / 64, DMODEL / 128), blk, 0, stream>>>(
                bufH, bufH,
                l2_w + (size_t)l * DFFN * DMODEL, l2_b + (size_t)l * DMODEL,
                bufC + (size_t)c0 * DMODEL, mv, DFFN, DMODEL, 0, 0);
        }
        // LN2: next = LN(xm + ffn) -> bufX (or d_out on final layer)
        float* dst = (l == NLAYERS - 1) ? (float*)d_out : bufX;
        resid_ln_kernel<<<dim3(TOK), blk, 0, stream>>>(
            bufB, bufC, n2_s + (size_t)l * DMODEL, n2_b + (size_t)l * DMODEL, dst, TOK);
        x = bufX;
    }
}
